// GIN_7404523618681
// MI455X (gfx1250) — compile-verified, matrix operand order checked by reading, not run on hardware
//
#include <hip/hip_runtime.h>
#include <cstdint>
#include <cstddef>

// ---------------------------------------------------------------------------
// Types for CDNA5 WMMA (wave32). v16bf = A/B operand (8 VGPRs), v8f = C/D.
// ---------------------------------------------------------------------------
typedef __bf16 bf16;
typedef bf16  v16bf __attribute__((ext_vector_type(16)));
typedef float v8f   __attribute__((ext_vector_type(8)));

// ---------------------------------------------------------------------------
// Weight pack: row-major f32 W[K][Nout] -> bf16 in WMMA B-operand layout.
// Packed element p: i = p&15, lane = (p>>4)&31, tile = p>>9, nt = tile%NT,
// kt = tile/NT.  K = kt*32 + (i/8)*16 + (lane/16)*8 + (i%8),  N = nt*16+lane%16
// (mirrors the 16-bit A-matrix lane layout from CDNA5 ISA §7.12.2).
// ---------------------------------------------------------------------------
__global__ void pack_w_kernel(const float* __restrict__ W, bf16* __restrict__ o,
                              int total, int Nout, int NT) {
    int p = blockIdx.x * blockDim.x + threadIdx.x;
    if (p >= total) return;
    int i    = p & 15;
    int lane = (p >> 4) & 31;
    int tile = p >> 9;
    int nt   = tile % NT;
    int kt   = tile / NT;
    int k = kt * 32 + ((i >> 3) << 4) + ((lane >> 4) << 3) + (i & 7);
    int n = nt * 16 + (lane & 15);
    o[p] = (bf16)W[(size_t)k * Nout + n];
}

// ---------------------------------------------------------------------------
// Elementwise residual: o = a + b   (N*128 floats, float4 vectorized)
// ---------------------------------------------------------------------------
__global__ void add_kernel(const float4* __restrict__ a, const float4* __restrict__ b,
                           float4* __restrict__ o, int n) {
    int i = blockIdx.x * blockDim.x + threadIdx.x;
    if (i < n) {
        float4 x = a[i], y = b[i];
        o[i] = make_float4(x.x + y.x, x.y + y.y, x.z + y.z, x.w + y.w);
    }
}

// ---------------------------------------------------------------------------
// Edge kernel: one wave per 16-edge strip.
//   el  = edge_feats[16x32] @ We[32x128]      (1 K-tile, 8 N-tiles of WMMA)
//   msg = relu(x[src] + el + be)              (gather; row-coalesced, hits L2)
//   agg[dst] += msg                           (global_atomic_add_f32, at L2)
// Epilogue is row-major: one base pointer per row, 8 immediate-offset column
// accesses -> minimal VALU address math, batched loads before the atomics.
// ---------------------------------------------------------------------------
__global__ void edge_kernel(const float* __restrict__ ef, const int* __restrict__ src,
                            const int* __restrict__ dst, const float* __restrict__ x,
                            const bf16* __restrict__ Wep, const float* __restrict__ be,
                            float* __restrict__ agg, int E) {
    const int lane = threadIdx.x & 31;
    const int wid  = blockIdx.x * (blockDim.x >> 5) + (threadIdx.x >> 5);
    const int e0   = wid * 16;
    if (e0 >= E) return;                      // wave-uniform: EXEC stays all-1s

    // A operand: 16 edge rows x 32 feats (bf16)
    v16bf a;
    {
        const int row = e0 + (lane & 15);
        const float* ar = ef + (size_t)row * 32;
        const int hi = (lane >> 4) << 3;
#pragma unroll
        for (int i = 0; i < 8; ++i) a[i]     = (bf16)ar[hi + i];
#pragma unroll
        for (int i = 0; i < 8; ++i) a[8 + i] = (bf16)ar[16 + hi + i];
    }

    const v16bf* Bt = (const v16bf*)Wep;
    v8f acc[8];
#pragma unroll
    for (int nt = 0; nt < 8; ++nt) {
        v8f c = {0.f, 0.f, 0.f, 0.f, 0.f, 0.f, 0.f, 0.f};
        v16bf b = Bt[nt * 32 + lane];
        acc[nt] = __builtin_amdgcn_wmma_f32_16x16x32_bf16(
            false, a, false, b, (short)0, c, false, false);
    }

    // D layout: lane l, reg r holds (edge = e0 + r + 8*(l/16), col = nt*16+l%16)
    const int mhi  = (lane >> 4) << 3;
    const unsigned ncol = (unsigned)(lane & 15);

    int se[8], de[8];
#pragma unroll
    for (int r = 0; r < 8; ++r) {
        const int e = e0 + mhi + r;
        se[r] = src[e];
        de[r] = dst[e];
    }

    float bb[8];
#pragma unroll
    for (int nt = 0; nt < 8; ++nt) bb[nt] = be[nt * 16 + ncol];

#pragma unroll
    for (int r = 0; r < 8; ++r) {
        const float* xrow = x   + ((unsigned)se[r] * 128u + ncol);
        float*       arow = agg + ((unsigned)de[r] * 128u + ncol);
        float xv[8];
#pragma unroll
        for (int nt = 0; nt < 8; ++nt) xv[nt] = xrow[nt * 16];   // imm offsets
#pragma unroll
        for (int nt = 0; nt < 8; ++nt) {
            float m = fmaxf(xv[nt] + acc[nt][r] + bb[nt], 0.f);
            unsafeAtomicAdd(&arow[nt * 16], m);                  // imm offsets
        }
    }
}

// ---------------------------------------------------------------------------
// Fused GEMM + bias + activation: out[M x NT*16] = act(A[M x KT*32] @ W + b)
// One wave per 16-row strip; NT accumulators (v8f) live in registers.
// ACT: 0 = none, 1 = relu, 2 = tanh
// ---------------------------------------------------------------------------
template <int KT, int NT, int ACT>
__global__ void gemm_kernel(const float* __restrict__ A, const bf16* __restrict__ Wp,
                            const float* __restrict__ bias, float* __restrict__ out,
                            int M) {
    const int lane = threadIdx.x & 31;
    const int wid  = blockIdx.x * (blockDim.x >> 5) + (threadIdx.x >> 5);
    const int m0   = wid * 16;
    if (m0 >= M) return;                      // wave-uniform

    constexpr int K    = KT * 32;
    constexpr int Nout = NT * 16;

    v8f acc[NT];
#pragma unroll
    for (int nt = 0; nt < NT; ++nt)
        acc[nt] = (v8f){0.f, 0.f, 0.f, 0.f, 0.f, 0.f, 0.f, 0.f};

    const int row = m0 + (lane & 15);
    const float* ar = A + (size_t)row * K;
    const int hi = (lane >> 4) << 3;
    const v16bf* Bt = (const v16bf*)Wp;

#pragma unroll
    for (int kt = 0; kt < KT; ++kt) {
        v16bf a;
        const int base = kt * 32 + hi;
#pragma unroll
        for (int i = 0; i < 8; ++i) a[i]     = (bf16)ar[base + i];
#pragma unroll
        for (int i = 0; i < 8; ++i) a[8 + i] = (bf16)ar[base + 16 + i];
#pragma unroll
        for (int nt = 0; nt < NT; ++nt) {
            v16bf b = Bt[(kt * NT + nt) * 32 + lane];
            acc[nt] = __builtin_amdgcn_wmma_f32_16x16x32_bf16(
                false, a, false, b, (short)0, acc[nt], false, false);
        }
    }

    const int mhi  = (lane >> 4) << 3;
    const unsigned ncol = (unsigned)(lane & 15);

    float bb[NT];
#pragma unroll
    for (int nt = 0; nt < NT; ++nt) bb[nt] = bias[nt * 16 + ncol];

#pragma unroll
    for (int r = 0; r < 8; ++r) {
        float* orow = out + ((unsigned)(m0 + mhi + r) * (unsigned)Nout + ncol);
#pragma unroll
        for (int nt = 0; nt < NT; ++nt) {
            float v = acc[nt][r] + bb[nt];
            if (ACT == 1) v = fmaxf(v, 0.f);
            else if (ACT == 2) v = tanhf(v);
            orow[nt * 16] = v;                                   // imm offsets
        }
    }
}

// ---------------------------------------------------------------------------
// Host-side orchestration (graph-capture safe: only async ops on `stream`).
// ---------------------------------------------------------------------------
extern "C" void kernel_launch(void* const* d_in, const int* in_sizes, int n_in,
                              void* d_out, int out_size, void* d_ws, size_t ws_size,
                              hipStream_t stream) {
    const float* x    = (const float*)d_in[0];
    const int*   eidx = (const int*)  d_in[1];
    const float* ef   = (const float*)d_in[2];
    const float* We1  = (const float*)d_in[3];
    const float* be1  = (const float*)d_in[4];
    const float* W1a  = (const float*)d_in[5];
    const float* b1a  = (const float*)d_in[6];
    const float* W1b  = (const float*)d_in[7];
    const float* b1b  = (const float*)d_in[8];
    const float* We2  = (const float*)d_in[9];
    const float* be2  = (const float*)d_in[10];
    const float* W2a  = (const float*)d_in[11];
    const float* b2a  = (const float*)d_in[12];
    const float* W2b  = (const float*)d_in[13];
    const float* b2b  = (const float*)d_in[14];
    const float* Wfc1 = (const float*)d_in[15];
    const float* bfc1 = (const float*)d_in[16];
    const float* Wfc2 = (const float*)d_in[17];
    const float* bfc2 = (const float*)d_in[18];

    const int N = in_sizes[0] / 128;   // 100000
    const int E = in_sizes[2] / 32;    // 600000
    const int* src = eidx;
    const int* dst = eidx + E;

    // Workspace layout
    const size_t NB = (size_t)N * 128 * sizeof(float);   // node buffer bytes
    char* ws = (char*)d_ws;
    float* n0  = (float*)(ws + 0 * NB);
    float* n1  = (float*)(ws + 1 * NB);
    float* n2  = (float*)(ws + 2 * NB);
    float* agg = (float*)(ws + 3 * NB);
    bf16* pw = (bf16*)(ws + 4 * NB);
    bf16* We1p = pw;           pw += 32 * 128;
    bf16* We2p = pw;           pw += 32 * 128;
    bf16* W1ap = pw;           pw += 128 * 128;
    bf16* W1bp = pw;           pw += 128 * 128;
    bf16* W2ap = pw;           pw += 128 * 128;
    bf16* W2bp = pw;           pw += 128 * 128;
    bf16* Wfc1p = pw;          pw += 128 * 128;
    bf16* Wfc2p = pw;          pw += 128 * 64;

    // ---- pack weights to bf16 WMMA B layout (tiny; weights stay L2-resident)
    auto pack = [&](const float* W, bf16* o, int K, int Nout) {
        int total = K * Nout;
        pack_w_kernel<<<(total + 255) / 256, 256, 0, stream>>>(W, o, total, Nout, Nout / 16);
    };
    pack(We1, We1p, 32, 128);
    pack(We2, We2p, 32, 128);
    pack(W1a, W1ap, 128, 128);
    pack(W1b, W1bp, 128, 128);
    pack(W2a, W2ap, 128, 128);
    pack(W2b, W2bp, 128, 128);
    pack(Wfc1, Wfc1p, 128, 128);
    pack(Wfc2, Wfc2p, 128, 64);

    const int wavesPerBlock = 4;                      // blockDim = 128
    const int strips_e = (E + 15) / 16;
    const int blocks_e = (strips_e + wavesPerBlock - 1) / wavesPerBlock;
    const int strips_n = (N + 15) / 16;
    const int blocks_n = (strips_n + wavesPerBlock - 1) / wavesPerBlock;
    const int addN = (int)(NB / sizeof(float4));
    const int blocks_a = (addN + 255) / 256;

    // ======================= conv1 =======================
    hipMemsetAsync(agg, 0, NB, stream);
    edge_kernel<<<blocks_e, 128, 0, stream>>>(ef, src, dst, x, We1p, be1, agg, E);
    add_kernel<<<blocks_a, 256, 0, stream>>>((const float4*)x, (const float4*)agg,
                                             (float4*)n0, addN);
    gemm_kernel<4, 8, 1><<<blocks_n, 128, 0, stream>>>(n0, W1ap, b1a, n1, N);
    gemm_kernel<4, 8, 2><<<blocks_n, 128, 0, stream>>>(n1, W1bp, b1b, n2, N); // h1 = n2

    // ======================= conv2 =======================
    hipMemsetAsync(agg, 0, NB, stream);
    edge_kernel<<<blocks_e, 128, 0, stream>>>(ef, src, dst, n2, We2p, be2, agg, E);
    add_kernel<<<blocks_a, 256, 0, stream>>>((const float4*)n2, (const float4*)agg,
                                             (float4*)n0, addN);
    gemm_kernel<4, 8, 1><<<blocks_n, 128, 0, stream>>>(n0, W2ap, b2a, n1, N);
    gemm_kernel<4, 8, 2><<<blocks_n, 128, 0, stream>>>(n1, W2bp, b2b, n0, N); // h2 = n0

    // ======================= heads =======================
    gemm_kernel<4, 8, 2><<<blocks_n, 128, 0, stream>>>(n0, Wfc1p, bfc1, n1, N);
    gemm_kernel<4, 4, 0><<<blocks_n, 128, 0, stream>>>(n1, Wfc2p, bfc2, (float*)d_out, N);
}
